// CISSVAE_38551626449609
// MI455X (gfx1250) — matrix operand order repaired and checked by reading, not compiled, not run on hardware
//
#include <hip/hip_runtime.h>

// ---------------- problem constants ----------------
#define BN   16384          // batch rows
#define DIN  512
#define LATD 64
#define NC   8
#define BP   16512          // BN + NC*16 padding (each cluster segment 16-aligned)
#define NT   (BP / 16)      // 1032 row tiles of 16

typedef __attribute__((ext_vector_type(16))) __bf16 v16bf;
typedef __attribute__((ext_vector_type(8)))  __bf16 v8bf;
typedef __attribute__((ext_vector_type(8)))  float  v8f;

__device__ __forceinline__ __bf16 f2bf(float f) {
    union { float f; unsigned u; } v; v.f = f;
    unsigned r = v.u + 0x7FFFu + ((v.u >> 16) & 1u);   // round-to-nearest-even
    unsigned short h = (unsigned short)(r >> 16);
    return __builtin_bit_cast(__bf16, h);
}
__device__ __forceinline__ float bf2f(__bf16 b) {
    unsigned short h = __builtin_bit_cast(unsigned short, b);
    union { unsigned u; float f; } v; v.u = ((unsigned)h) << 16;
    return v.f;
}

// ---------------- setup kernels ----------------
__global__ void k_init(int* counts, int* fill, int* tile_cluster, int* perm) {
    int i = blockIdx.x * blockDim.x + threadIdx.x;
    if (i < NC) { counts[i] = 0; fill[i] = 0; }
    if (i < NT) tile_cluster[i] = -1;
    if (i < BP) perm[i] = -1;
}

__global__ void k_count(const int* __restrict__ labels, int* counts) {
    int i = blockIdx.x * blockDim.x + threadIdx.x;
    if (i < BN) atomicAdd(&counts[labels[i]], 1);
}

__global__ void k_scan(const int* __restrict__ counts, int* seg_start, int* fill,
                       int* tile_cluster) {
    if (threadIdx.x == 0 && blockIdx.x == 0) {
        int off = 0;
        for (int c = 0; c < NC; ++c) {
            seg_start[c] = off;
            fill[c] = off;
            int tiles = (counts[c] + 15) >> 4;
            int t0 = off >> 4;
            for (int t = 0; t < tiles; ++t) tile_cluster[t0 + t] = c;
            off += tiles << 4;
        }
        seg_start[NC] = off;
    }
}

__global__ void k_place(const int* __restrict__ labels, int* fill, int* perm) {
    int i = blockIdx.x * blockDim.x + threadIdx.x;
    if (i < BN) {
        int c = labels[i];
        int pos = atomicAdd(&fill[c], 1);
        perm[pos] = i;
    }
}

// transpose-convert weights: src f32 [C][K][N] -> dst bf16 [C][N][K]
__global__ void k_convT(const float* __restrict__ src, __bf16* __restrict__ dst,
                        int K, int N, int C) {
    long long total = (long long)C * K * N;
    long long i = (long long)blockIdx.x * blockDim.x + threadIdx.x;
    if (i >= total) return;
    int c = (int)(i / ((long long)K * N));
    int rem = (int)(i - (long long)c * K * N);
    int k = rem / N;
    int n = rem - k * N;
    dst[(long long)c * K * N + (long long)n * K + k] = f2bf(src[i]);
}

__global__ void k_bias_cat(const float* __restrict__ mu_b, const float* __restrict__ lv_b,
                           float* __restrict__ out) {
    int i = threadIdx.x;
    if (i < LATD) out[i] = mu_b[i];
    else if (i < 2 * LATD) out[i] = lv_b[i - LATD];
}

// gather x rows into sorted+padded bf16 activation buffer
__global__ void k_gather_x(const float* __restrict__ x, const int* __restrict__ perm,
                           __bf16* __restrict__ A0) {
    long long i = (long long)blockIdx.x * blockDim.x + threadIdx.x;
    if (i >= (long long)BP * DIN) return;
    int p = (int)(i / DIN);
    int k = (int)(i - (long long)p * DIN);
    int r = perm[p];
    float v = (r >= 0) ? x[(long long)r * DIN + k] : 0.f;
    A0[i] = f2bf(v);
}

// z = mu + exp(0.5*logvar) * eps   (mu|lv packed as [BP][128], eps gathered via perm)
__global__ void k_reparam(const __bf16* __restrict__ mulv, const float* __restrict__ eps,
                          const int* __restrict__ perm, __bf16* __restrict__ Z) {
    long long i = (long long)blockIdx.x * blockDim.x + threadIdx.x;
    if (i >= (long long)BP * LATD) return;
    int p = (int)(i / LATD);
    int j = (int)(i - (long long)p * LATD);
    float mu = bf2f(mulv[(long long)p * 128 + j]);
    float lv = bf2f(mulv[(long long)p * 128 + LATD + j]);
    int r = perm[p];
    float e = (r >= 0) ? eps[(long long)r * LATD + j] : 0.f;
    Z[i] = f2bf(mu + __expf(0.5f * lv) * e);
}

// ---------------- WMMA GEMM (register-blocked along N, pipelined B loads) ----------------
// A: bf16 [BP][K] sorted rows. Wt: bf16 [C][N][K] pre-transposed. bias: f32 [C][N].
// One block (8 waves) handles rowTile = blockIdx.x and ALL N columns:
// wave w owns CT = N/128 output tiles (colTile = j*8 + w). Per K-step of 32,
// ALL CT B-fragments are loaded into distinct registers first, then the CT
// WMMAs drain with descending partial loadcnt waits (loads of later j and of
// the next K-step overlap with earlier WMMAs).
template <int CT, bool RELU, bool FINAL>
__global__ __launch_bounds__(256, 2)
void wmma_gemm(const __bf16* __restrict__ A, int K,
               const __bf16* __restrict__ Wt, const float* __restrict__ bias, int N,
               long long wStride, int bStride,
               const int* __restrict__ tile_cluster, const int* __restrict__ perm,
               __bf16* __restrict__ Y, float* __restrict__ Yf) {
    int rowTile = blockIdx.x;
    int c = tile_cluster[rowTile];
    if (c < 0) return;                                  // wholly-padding tile (block-uniform)

    int wave = threadIdx.x >> 5;
    int lane = threadIdx.x & 31;

    const __bf16* W  = Wt   + (long long)c * wStride;
    const float*  bp = bias + (long long)c * bStride;

    int m  = lane & 15;
    int kh = lane >> 4;
    const __bf16* Arow  = A + (long long)(rowTile * 16 + m) * K;       // A frag: M = lane&15
    const __bf16* Wcol0 = W + (long long)(wave * 16 + m) * K;          // B frag j=0: N = lane&15
    const long long wjStride = (long long)128 * K;                     // colTile j stride

    v8f acc[CT];
#pragma unroll
    for (int j = 0; j < CT; ++j) acc[j] = (v8f){};

    for (int k0 = 0; k0 < K; k0 += 32) {
        // A 16x32 bf16 layout: lanes0-15 K=kh*8+{0..7} then 16+kh*8+{0..7}
        v8bf alo = *(const v8bf*)(Arow + k0 + kh * 8);
        v8bf ahi = *(const v8bf*)(Arow + k0 + 16 + kh * 8);
        __builtin_prefetch(Arow + k0 + 128, 0, 3);
        v16bf av = __builtin_shufflevector(alo, ahi, 0,1,2,3,4,5,6,7,8,9,10,11,12,13,14,15);

        // stage ALL B fragments for this K-step (distinct regs -> one load clause)
        v16bf bv[CT];
#pragma unroll
        for (int j = 0; j < CT; ++j) {
            // B 32x16 bf16 layout: lanes0-15 K=kh*8+{0..7} then 16+kh*8+{0..7}
            const __bf16* wp = Wcol0 + (long long)j * wjStride + k0 + kh * 8;
            v8bf blo = *(const v8bf*)(wp);
            v8bf bhi = *(const v8bf*)(wp + 16);
            bv[j] = __builtin_shufflevector(blo, bhi, 0,1,2,3,4,5,6,7,8,9,10,11,12,13,14,15);
        }
        // drain WMMAs (wait only for the fragment actually consumed)
#pragma unroll
        for (int j = 0; j < CT; ++j) {
            acc[j] = __builtin_amdgcn_wmma_f32_16x16x32_bf16(false, av, false, bv[j],
                                                             (short)0, acc[j], false, false);
        }
    }

    // C/D layout: lanes0-15 -> M=r, N=lane ; lanes16-31 -> M=8+r, N=lane-16
    int rbase = rowTile * 16 + (lane >> 4) * 8;
#pragma unroll
    for (int j = 0; j < CT; ++j) {
        int col = (j * 8 + wave) * 16 + (lane & 15);
        float bv = bp[col];
#pragma unroll
        for (int r = 0; r < 8; ++r) {
            float y = acc[j][r] + bv;
            if (RELU) y = fmaxf(y, 0.f);
            int p = rbase + r;
            if (FINAL) {
                int orig = perm[p];
                if (orig >= 0) Yf[(long long)orig * N + col] = y;
            } else {
                Y[(long long)p * N + col] = f2bf(y);
            }
        }
    }
}

// ---------------- host launcher ----------------
extern "C" void kernel_launch(void* const* d_in, const int* in_sizes, int n_in,
                              void* d_out, int out_size, void* d_ws, size_t ws_size,
                              hipStream_t stream) {
    const float* x       = (const float*)d_in[0];
    const int*   labels  = (const int*)  d_in[1];
    const float* eps     = (const float*)d_in[2];
    const float* enc_W0  = (const float*)d_in[3];
    const float* enc_b0  = (const float*)d_in[4];
    const float* enc_Wu  = (const float*)d_in[5];
    const float* enc_bu  = (const float*)d_in[6];
    const float* enc_W2  = (const float*)d_in[7];
    const float* enc_b2  = (const float*)d_in[8];
    const float* mu_W    = (const float*)d_in[9];
    const float* mu_b    = (const float*)d_in[10];
    const float* lv_W    = (const float*)d_in[11];
    const float* lv_b    = (const float*)d_in[12];
    const float* dec_Wu0 = (const float*)d_in[13];
    const float* dec_bu0 = (const float*)d_in[14];
    const float* dec_W1  = (const float*)d_in[15];
    const float* dec_b1  = (const float*)d_in[16];
    const float* dec_Wu2 = (const float*)d_in[17];
    const float* dec_bu2 = (const float*)d_in[18];
    const float* fin_W   = (const float*)d_in[19];
    const float* fin_b   = (const float*)d_in[20];
    float* out = (float*)d_out;

    // bump allocator over d_ws (256B aligned partitions)
    size_t off = 0;
    auto alloc = [&](size_t bytes) -> void* {
        off = (off + 255) & ~(size_t)255;
        void* p = (char*)d_ws + off;
        off += bytes;
        return p;
    };
    int* counts  = (int*)alloc(NC * 4);
    int* segs    = (int*)alloc((NC + 1) * 4);
    int* fill    = (int*)alloc(NC * 4);
    int* tilec   = (int*)alloc(NT * 4);
    int* perm    = (int*)alloc(BP * 4);

    __bf16* Wt_enc0 = (__bf16*)alloc((size_t)512 * 1024 * 2);
    __bf16* Wt_encU = (__bf16*)alloc((size_t)NC * 1024 * 512 * 2);
    __bf16* Wt_enc2 = (__bf16*)alloc((size_t)512 * 256 * 2);
    __bf16* Wt_mulv = (__bf16*)alloc((size_t)256 * 128 * 2);
    __bf16* Wt_dec0 = (__bf16*)alloc((size_t)NC * 64 * 256 * 2);
    __bf16* Wt_dec1 = (__bf16*)alloc((size_t)256 * 512 * 2);
    __bf16* Wt_dec2 = (__bf16*)alloc((size_t)NC * 512 * 1024 * 2);
    __bf16* Wt_fin  = (__bf16*)alloc((size_t)NC * 1024 * 512 * 2);
    float*  b_mulv  = (float*)alloc(128 * 4);

    __bf16* bufA = (__bf16*)alloc((size_t)BP * 1024 * 2);
    __bf16* bufB = (__bf16*)alloc((size_t)BP * 1024 * 2);

    // ---- setup: sort rows by cluster ----
    k_init <<<(BP + 255) / 256, 256, 0, stream>>>(counts, fill, tilec, perm);
    k_count<<<(BN + 255) / 256, 256, 0, stream>>>(labels, counts);
    k_scan <<<1, 32, 0, stream>>>(counts, segs, fill, tilec);
    k_place<<<(BN + 255) / 256, 256, 0, stream>>>(labels, fill, perm);

    // ---- weight transpose-convert to bf16 [N][K] ----
    auto conv = [&](const float* s, __bf16* d, int K, int N, int C) {
        long long tot = (long long)C * K * N;
        k_convT<<<(unsigned)((tot + 255) / 256), 256, 0, stream>>>(s, d, K, N, C);
    };
    conv(enc_W0,  Wt_enc0, 512, 1024, 1);
    conv(enc_Wu,  Wt_encU, 1024, 512, NC);
    conv(enc_W2,  Wt_enc2, 512, 256, 1);
    conv(mu_W,    Wt_mulv,             256, 64, 1);     // cols 0..63
    conv(lv_W,    Wt_mulv + 64 * 256,  256, 64, 1);     // cols 64..127
    conv(dec_Wu0, Wt_dec0, 64, 256, NC);
    conv(dec_W1,  Wt_dec1, 256, 512, 1);
    conv(dec_Wu2, Wt_dec2, 512, 1024, NC);
    conv(fin_W,   Wt_fin,  1024, 512, NC);
    k_bias_cat<<<1, 128, 0, stream>>>(mu_b, lv_b, b_mulv);

    // ---- gather x (sorted, padded, bf16) ----
    k_gather_x<<<(unsigned)(((long long)BP * DIN + 255) / 256), 256, 0, stream>>>(x, perm, bufA);

    // ---- GEMM chain (grid = NT row-tiles; block covers full N) ----
    // encoder: 512 -> 1024 (relu)
    wmma_gemm<8, true, false><<<NT, 256, 0, stream>>>(bufA, 512, Wt_enc0, enc_b0, 1024,
        0, 0, tilec, perm, bufB, nullptr);
    // unshared: 1024 -> 512 (relu, per-cluster)
    wmma_gemm<4, true, false><<<NT, 256, 0, stream>>>(bufB, 1024, Wt_encU, enc_bu, 512,
        (long long)1024 * 512, 512, tilec, perm, bufA, nullptr);
    // 512 -> 256 (relu)
    wmma_gemm<2, true, false><<<NT, 256, 0, stream>>>(bufA, 512, Wt_enc2, enc_b2, 256,
        0, 0, tilec, perm, bufB, nullptr);
    // fused mu|logvar heads: 256 -> 128 (no activation)
    wmma_gemm<1, false, false><<<NT, 256, 0, stream>>>(bufB, 256, Wt_mulv, b_mulv, 128,
        0, 0, tilec, perm, bufA, nullptr);
    // reparameterize -> Z bf16 [BP][64]
    k_reparam<<<(unsigned)(((long long)BP * LATD + 255) / 256), 256, 0, stream>>>(bufA, eps, perm, bufB);
    // unshared: 64 -> 256 (relu, per-cluster)
    wmma_gemm<2, true, false><<<NT, 256, 0, stream>>>(bufB, 64, Wt_dec0, dec_bu0, 256,
        (long long)64 * 256, 256, tilec, perm, bufA, nullptr);
    // 256 -> 512 (relu)
    wmma_gemm<4, true, false><<<NT, 256, 0, stream>>>(bufA, 256, Wt_dec1, dec_b1, 512,
        0, 0, tilec, perm, bufB, nullptr);
    // unshared: 512 -> 1024 (relu, per-cluster)
    wmma_gemm<8, true, false><<<NT, 256, 0, stream>>>(bufB, 512, Wt_dec2, dec_bu2, 1024,
        (long long)512 * 1024, 1024, tilec, perm, bufA, nullptr);
    // final per-cluster layer 1024 -> 512, f32 scatter to original row order
    wmma_gemm<4, false, true><<<NT, 256, 0, stream>>>(bufA, 1024, Wt_fin, fin_b, 512,
        (long long)1024 * 512, 512, tilec, perm, nullptr, out);
}
